// MultiHeadAttention_72327249265052
// MI455X (gfx1250) — matrix operand order repaired and checked by reading, not compiled, not run on hardware
//
#include <hip/hip_runtime.h>
#include <hip/hip_bf16.h>

// ---------------------------------------------------------------------------
// MHA block for MI455X (gfx1250, wave32, WMMA bf16 16x16x32, TDM staging).
//   B=4, S=2048, D=1024, H=16, HD=64
// Pipeline:
//   0) cvt_kernel  : fp32 -> bf16 for Q,K,V and Wq,Wk,Wv,Wo (bandwidth pass)
//   1) proj_gemm   : TDM double-buffered LDS tiles, bf16 WMMA
//                    q = (Q Wq^T)*0.125, k = K Wk^T -> [B,H,S,HD] bf16
//                    vT = (V Wv^T)^T                -> [B,H,HD,S] bf16
//   2) attn_kernel : flash attention, per-wave 16-row q tile, bf16 WMMA
//   3) out_gemm    : TDM-staged attn @ Wo^T + residual(Q) -> f32
//   4) ln_kernel   : LayerNorm(eps=1e-6) -> d_out
// Workspace (needs 120 MB):
//   [  0, 16M) Qbf   (reused later as f32 pre-LN buffer [0,32M))
//   [ 16, 32M) Kbf
//   [ 32, 48M) Vbf
//   [ 48, 56M) Wq/Wk/Wv/Wo bf16 (2MB each)
//   [ 56, 72M) q   [ 72, 88M) k   [ 88,104M) vT   [104,120M) attn_out
// ---------------------------------------------------------------------------

typedef __bf16 bf16;
typedef __attribute__((ext_vector_type(8)))  bf16  bf16x8;
typedef __attribute__((ext_vector_type(16))) bf16  v16bf;
typedef __attribute__((ext_vector_type(8)))  float f32x8;
typedef __attribute__((ext_vector_type(8)))  float v8f;
typedef __attribute__((ext_vector_type(4)))  unsigned int u32x4;
typedef __attribute__((ext_vector_type(4)))  int   i32x4;
typedef __attribute__((ext_vector_type(8)))  int   i32x8;

#define Bc 4
#define Sc 2048
#define Dc 1024
#define Hc 16
#define HDc 64

static __device__ inline v8f wmma_bf16(v16bf a, v16bf b, v8f c) {
  return __builtin_amdgcn_wmma_f32_16x16x32_bf16(false, a, false, b,
                                                 (short)0, c, false, false);
}

// ---- A fragment (16x32) from row-major bf16 [rows x ld]
// lane m (0..15) holds row m: K = {hi*8..+7, 16+hi*8..+7}, hi = lane/16
static __device__ inline v16bf a_frag_bf16(const bf16* A, int ld, int row0,
                                           int k0, int lane) {
  const int m = lane & 15, hi = lane >> 4;
  const bf16* r = A + (size_t)(row0 + m) * ld;
  bf16x8 c0 = *(const bf16x8*)(r + k0 + hi * 8);
  bf16x8 c1 = *(const bf16x8*)(r + k0 + 16 + hi * 8);
  v16bf a;
#pragma unroll
  for (int i = 0; i < 8; ++i) { a[i] = c0[i]; a[i + 8] = c1[i]; }
  return a;
}

// ---- B fragment (32x16): B[k][n] = M[n0+n][k0+k], row-major bf16 [N x ld]
// lane n (0..15) holds column n: K = {hi*8..+7, 16+hi*8..+7}, hi = lane/16
static __device__ inline v16bf b_frag_bf16(const bf16* M, int ld, int n0,
                                           int k0, int lane) {
  const int n = lane & 15, hi = lane >> 4;
  const bf16* r = M + (size_t)(n0 + n) * ld + k0 + hi * 8;
  bf16x8 c0 = *(const bf16x8*)(r);
  bf16x8 c1 = *(const bf16x8*)(r + 16);
  v16bf b;
#pragma unroll
  for (int i = 0; i < 8; ++i) { b[i] = c0[i]; b[i + 8] = c1[i]; }
  return b;
}

// ---------------------------------------------------------------------------
// Tensor Data Mover: 2D tile (rows x cols, bf16) from global -> LDS.
// D# built per CDNA5 ISA (count=1, data_size=2B,
// type=2). Groups 2/3 zero (2D tensor). Tracked on TENSORcnt.
// ---------------------------------------------------------------------------
static __device__ inline void tdm_load_2d(unsigned int lds_off,
                                          const bf16* gbase,
                                          unsigned long long row0,
                                          unsigned long long col0,
                                          unsigned int rows, unsigned int cols,
                                          unsigned int tensor_cols,
                                          unsigned int tensor_rows,
                                          unsigned long long stride_elems) {
  const unsigned long long gaddr =
      (unsigned long long)(uintptr_t)gbase + (row0 * stride_elems + col0) * 2ull;
  u32x4 g0;
  g0[0] = 1u;                                  // count=1, user mode
  g0[1] = lds_off;                             // lds_addr
  g0[2] = (unsigned int)(gaddr & 0xffffffffu); // global_addr[31:0]
  g0[3] = (unsigned int)((gaddr >> 32) & 0x01ffffffu) | (2u << 30);  // type=2
  i32x8 g1;
  g1[0] = (int)(1u << 16);                     // data_size = 1 (2 bytes)
  g1[1] = (int)((tensor_cols & 0xffffu) << 16);            // dim0[15:0]
  g1[2] = (int)((tensor_cols >> 16) & 0xffffu) |
          (int)((tensor_rows & 0xffffu) << 16);            // dim0[31:16]|dim1
  g1[3] = (int)((tensor_rows >> 16) & 0xffffu) |
          (int)((cols & 0xffffu) << 16);                   // dim1[31:16]|tile0
  g1[4] = (int)(rows & 0xffffu);                           // tile1 (tile2=0)
  g1[5] = (int)(stride_elems & 0xffffffffu);               // dim0_stride[31:0]
  g1[6] = (int)((stride_elems >> 32) & 0xffffu);           // dim0_stride[47:32]
  g1[7] = 0;
  const i32x4 z4 = {0, 0, 0, 0};
#if defined(__clang_major__) && (__clang_major__ >= 23)
  const i32x8 z8 = {0, 0, 0, 0, 0, 0, 0, 0};
  __builtin_amdgcn_tensor_load_to_lds(g0, g1, z4, z4, z8, 0);
#else
  __builtin_amdgcn_tensor_load_to_lds(g0, g1, z4, z4, 0);
#endif
}

static __device__ inline unsigned int lds_offset_of(const void* p) {
  return (unsigned int)(uintptr_t)p;  // LDS aperture: offset = addr[31:0]
}

// ===========================================================================
// Kernel 0: fp32 -> bf16 conversion, 8 elems/thread.
// ===========================================================================
__global__ __launch_bounds__(256) void cvt_kernel(const float* __restrict__ s,
                                                  bf16* __restrict__ d) {
  const size_t i = ((size_t)blockIdx.x * blockDim.x + threadIdx.x) * 8;
  f32x8 v = *(const f32x8*)(s + i);
  bf16x8 o;
#pragma unroll
  for (int j = 0; j < 8; ++j) o[j] = (bf16)v[j];
  *(bf16x8*)(d + i) = o;
}

// ===========================================================================
// Kernel 1: fused QKV projection. grid=(64,8,3), block=256 (8 waves).
// Per-block 128x128 tile; TDM stages X-tile and W-tile (128x32 bf16 each)
// into double-buffered LDS; wave (4x2) computes 32x64 = 2x4 WMMA tiles.
// ===========================================================================
__global__ __launch_bounds__(256) void proj_gemm(
    const bf16* __restrict__ Xq, const bf16* __restrict__ Xk,
    const bf16* __restrict__ Xv, const bf16* __restrict__ Wq,
    const bf16* __restrict__ Wk, const bf16* __restrict__ Wv,
    bf16* __restrict__ qo, bf16* __restrict__ ko, bf16* __restrict__ vTo) {
  const int z = blockIdx.z;
  const bf16* X = (z == 0) ? Xq : (z == 1) ? Xk : Xv;
  const bf16* W = (z == 0) ? Wq : (z == 1) ? Wk : Wv;

  const int wave = threadIdx.x >> 5, lane = threadIdx.x & 31;
  const int wm = wave & 3, wn = wave >> 2;
  const int m_blk = blockIdx.x * 128;
  const int n_blk = blockIdx.y * 128;

  __shared__ __align__(16) bf16 xtile[2][128 * 32];
  __shared__ __align__(16) bf16 wtile[2][128 * 32];

  v8f acc[2][4];
#pragma unroll
  for (int tm = 0; tm < 2; ++tm)
#pragma unroll
    for (int tn = 0; tn < 4; ++tn)
#pragma unroll
      for (int i = 0; i < 8; ++i) acc[tm][tn][i] = 0.f;

  if (wave == 0) {  // prologue: fill buffer 0 (TDM ignores EXEC; 1 issue/wave)
    tdm_load_2d(lds_offset_of(&xtile[0][0]), X, (unsigned)m_blk, 0u, 128, 32,
                Dc, Bc * Sc, Dc);
    tdm_load_2d(lds_offset_of(&wtile[0][0]), W, (unsigned)n_blk, 0u, 128, 32,
                Dc, Dc, Dc);
  }

  for (int k0 = 0; k0 < Dc; k0 += 32) {
    const int cur = (k0 >> 5) & 1;
    if (wave == 0) __builtin_amdgcn_s_wait_tensorcnt(0);
    __syncthreads();  // buffer `cur` ready; prev reads of `cur^1` finished
    if (wave == 0 && (k0 + 32) < Dc) {
      tdm_load_2d(lds_offset_of(&xtile[cur ^ 1][0]), X, (unsigned)m_blk,
                  (unsigned)(k0 + 32), 128, 32, Dc, Bc * Sc, Dc);
      tdm_load_2d(lds_offset_of(&wtile[cur ^ 1][0]), W, (unsigned)n_blk,
                  (unsigned)(k0 + 32), 128, 32, Dc, Dc, Dc);
    }
    v16bf a[2];
#pragma unroll
    for (int tm = 0; tm < 2; ++tm)
      a[tm] = a_frag_bf16(&xtile[cur][0], 32, wm * 32 + tm * 16, 0, lane);
    v16bf b[4];
#pragma unroll
    for (int tn = 0; tn < 4; ++tn)
      b[tn] = b_frag_bf16(&wtile[cur][0], 32, wn * 64 + tn * 16, 0, lane);
#pragma unroll
    for (int tm = 0; tm < 2; ++tm)
#pragma unroll
      for (int tn = 0; tn < 4; ++tn)
        acc[tm][tn] = wmma_bf16(a[tm], b[tn], acc[tm][tn]);
  }

  const float scale = (z == 0) ? 0.125f : 1.0f;  // fold 1/sqrt(HD) into q
  const int hi = lane >> 4, nl = lane & 15;
#pragma unroll
  for (int tm = 0; tm < 2; ++tm)
#pragma unroll
    for (int tn = 0; tn < 4; ++tn)
#pragma unroll
      for (int r = 0; r < 8; ++r) {
        const int m = m_blk + wm * 32 + tm * 16 + r + hi * 8;  // b*S + s
        const int n = n_blk + wn * 64 + tn * 16 + nl;          // h*HD + hd
        const int bb = m >> 11, s = m & (Sc - 1);
        const int h = n >> 6, hd = n & (HDc - 1);
        const bf16 val = (bf16)(acc[tm][tn][r] * scale);
        if (z < 2) {
          bf16* dst = (z == 0) ? qo : ko;
          dst[(((size_t)(bb * Hc + h)) * Sc + s) * HDc + hd] = val;
        } else {
          vTo[(((size_t)(bb * Hc + h)) * HDc + hd) * Sc + s] = val;
        }
      }
}

// ===========================================================================
// Kernel 2: flash attention. grid = B*H*(S/64) = 2048, block = 128 (4 waves).
// Each wave owns one 16-row query tile; loops over keys in blocks of 32.
// K/V panels (512 KB/head) stay L2-resident on the 192 MB L2.
// ===========================================================================
__global__ __launch_bounds__(128) void attn_kernel(
    const bf16* __restrict__ q, const bf16* __restrict__ k,
    const bf16* __restrict__ vT, bf16* __restrict__ ao) {
  const int wave = threadIdx.x >> 5, lane = threadIdx.x & 31;
  const int qblk = blockIdx.x & (Sc / 64 - 1);
  const int bh = blockIdx.x >> 5;  // b*H + h
  const int s0 = qblk * 64 + wave * 16;

  const bf16* qp = q + (size_t)bh * Sc * HDc;
  const bf16* kp = k + (size_t)bh * Sc * HDc;
  const bf16* vp = vT + (size_t)bh * HDc * Sc;

  const v16bf qa0 = a_frag_bf16(qp, HDc, s0, 0, lane);
  const v16bf qa1 = a_frag_bf16(qp, HDc, s0, 32, lane);

  v8f o[4];
#pragma unroll
  for (int t = 0; t < 4; ++t)
#pragma unroll
    for (int i = 0; i < 8; ++i) o[t][i] = 0.f;

  float rowmax[8], rowsum[8];
#pragma unroll
  for (int r = 0; r < 8; ++r) { rowmax[r] = -1e30f; rowsum[r] = 0.f; }

  // Per-wave 16x32 bf16 P-tile: C-layout -> A-layout transpose through LDS.
  __shared__ __align__(16) bf16 pbuf[4][16 * 32];
  bf16* myp = pbuf[wave];
  const int hi = lane >> 4, nl = lane & 15;

  for (int j = 0; j < Sc; j += 32) {
    v8f sc[2];
#pragma unroll
    for (int t = 0; t < 2; ++t) {
#pragma unroll
      for (int i = 0; i < 8; ++i) sc[t][i] = 0.f;
      const v16bf kb0 = b_frag_bf16(kp, HDc, j + t * 16, 0, lane);
      const v16bf kb1 = b_frag_bf16(kp, HDc, j + t * 16, 32, lane);
      sc[t] = wmma_bf16(qa0, kb0, sc[t]);
      sc[t] = wmma_bf16(qa1, kb1, sc[t]);
    }

#pragma unroll
    for (int r = 0; r < 8; ++r) {
      float mblk = fmaxf(sc[0][r], sc[1][r]);
#pragma unroll
      for (int off = 8; off >= 1; off >>= 1)
        mblk = fmaxf(mblk, __shfl_xor(mblk, off, 16));
      const float mnew = fmaxf(rowmax[r], mblk);
      const float corr = __expf(rowmax[r] - mnew);
      rowmax[r] = mnew;
      const float p0 = __expf(sc[0][r] - mnew);
      const float p1 = __expf(sc[1][r] - mnew);
      float ps = p0 + p1;
#pragma unroll
      for (int off = 8; off >= 1; off >>= 1) ps += __shfl_xor(ps, off, 16);
      rowsum[r] = rowsum[r] * corr + ps;
#pragma unroll
      for (int t = 0; t < 4; ++t) o[t][r] *= corr;
      const int row = r + hi * 8;
      myp[row * 32 + nl] = (bf16)p0;        // same-wave LDS ops are in-order:
      myp[row * 32 + 16 + nl] = (bf16)p1;   // no barrier needed
    }

    const v16bf pa = a_frag_bf16(myp, 32, 0, 0, lane);
#pragma unroll
    for (int t = 0; t < 4; ++t) {
      const v16bf vb = b_frag_bf16(vp, Sc, t * 16, j, lane);
      o[t] = wmma_bf16(pa, vb, o[t]);
    }
  }

  const int bb = bh >> 4, h = bh & 15;
#pragma unroll
  for (int r = 0; r < 8; ++r) {
    const float inv = 1.f / rowsum[r];
    const int s = s0 + r + hi * 8;
#pragma unroll
    for (int t = 0; t < 4; ++t) {
      const int d = h * HDc + t * 16 + nl;
      ao[((size_t)(bb * Sc + s)) * Dc + d] = (bf16)(o[t][r] * inv);
    }
  }
}

// ===========================================================================
// Kernel 3: output projection + residual.  grid=(64,8), block=256. TDM-staged.
// ===========================================================================
__global__ __launch_bounds__(256) void out_gemm(
    const bf16* __restrict__ A, const bf16* __restrict__ Wo,
    const float* __restrict__ resid, float* __restrict__ out) {
  const int wave = threadIdx.x >> 5, lane = threadIdx.x & 31;
  const int wm = wave & 3, wn = wave >> 2;
  const int m_blk = blockIdx.x * 128;
  const int n_blk = blockIdx.y * 128;

  __shared__ __align__(16) bf16 xtile[2][128 * 32];
  __shared__ __align__(16) bf16 wtile[2][128 * 32];

  v8f acc[2][4];
#pragma unroll
  for (int tm = 0; tm < 2; ++tm)
#pragma unroll
    for (int tn = 0; tn < 4; ++tn)
#pragma unroll
      for (int i = 0; i < 8; ++i) acc[tm][tn][i] = 0.f;

  if (wave == 0) {
    tdm_load_2d(lds_offset_of(&xtile[0][0]), A, (unsigned)m_blk, 0u, 128, 32,
                Dc, Bc * Sc, Dc);
    tdm_load_2d(lds_offset_of(&wtile[0][0]), Wo, (unsigned)n_blk, 0u, 128, 32,
                Dc, Dc, Dc);
  }

  for (int k0 = 0; k0 < Dc; k0 += 32) {
    const int cur = (k0 >> 5) & 1;
    if (wave == 0) __builtin_amdgcn_s_wait_tensorcnt(0);
    __syncthreads();
    if (wave == 0 && (k0 + 32) < Dc) {
      tdm_load_2d(lds_offset_of(&xtile[cur ^ 1][0]), A, (unsigned)m_blk,
                  (unsigned)(k0 + 32), 128, 32, Dc, Bc * Sc, Dc);
      tdm_load_2d(lds_offset_of(&wtile[cur ^ 1][0]), Wo, (unsigned)n_blk,
                  (unsigned)(k0 + 32), 128, 32, Dc, Dc, Dc);
    }
    v16bf a[2];
#pragma unroll
    for (int tm = 0; tm < 2; ++tm)
      a[tm] = a_frag_bf16(&xtile[cur][0], 32, wm * 32 + tm * 16, 0, lane);
    v16bf b[4];
#pragma unroll
    for (int tn = 0; tn < 4; ++tn)
      b[tn] = b_frag_bf16(&wtile[cur][0], 32, wn * 64 + tn * 16, 0, lane);
#pragma unroll
    for (int tm = 0; tm < 2; ++tm)
#pragma unroll
      for (int tn = 0; tn < 4; ++tn)
        acc[tm][tn] = wmma_bf16(a[tm], b[tn], acc[tm][tn]);
  }

  const int hi = lane >> 4, nl = lane & 15;
#pragma unroll
  for (int tm = 0; tm < 2; ++tm)
#pragma unroll
    for (int tn = 0; tn < 4; ++tn)
#pragma unroll
      for (int r = 0; r < 8; ++r) {
        const int m = m_blk + wm * 32 + tm * 16 + r + hi * 8;
        const int n = n_blk + wn * 64 + tn * 16 + nl;
        const size_t idx = (size_t)m * Dc + n;
        out[idx] = acc[tm][tn][r] + resid[idx];
      }
}

// ===========================================================================
// Kernel 4: row LayerNorm (eps=1e-6).  grid=8192 rows, block=256.
// ===========================================================================
__global__ __launch_bounds__(256) void ln_kernel(
    const float* __restrict__ x, const float* __restrict__ g,
    const float* __restrict__ bta, float* __restrict__ y) {
  const int row = blockIdx.x;
  const int c = threadIdx.x * 4;
  const float* xr = x + (size_t)row * Dc;
  const float4 xv = *(const float4*)(xr + c);
  float s = xv.x + xv.y + xv.z + xv.w;
  float q = xv.x * xv.x + xv.y * xv.y + xv.z * xv.z + xv.w * xv.w;
#pragma unroll
  for (int off = 16; off >= 1; off >>= 1) {
    s += __shfl_xor(s, off, 32);
    q += __shfl_xor(q, off, 32);
  }
  __shared__ float ss[8], sq[8];
  const int wv = threadIdx.x >> 5, ln = threadIdx.x & 31;
  if (ln == 0) { ss[wv] = s; sq[wv] = q; }
  __syncthreads();
  float ts = 0.f, tq = 0.f;
#pragma unroll
  for (int i = 0; i < 8; ++i) { ts += ss[i]; tq += sq[i]; }
  const float mean = ts * (1.f / Dc);
  const float var = tq * (1.f / Dc) - mean * mean;
  const float rstd = rsqrtf(var + 1e-6f);
  const float4 gv = *(const float4*)(g + c);
  const float4 bv = *(const float4*)(bta + c);
  float4 yv;
  yv.x = (xv.x - mean) * rstd * gv.x + bv.x;
  yv.y = (xv.y - mean) * rstd * gv.y + bv.y;
  yv.z = (xv.z - mean) * rstd * gv.z + bv.z;
  yv.w = (xv.w - mean) * rstd * gv.w + bv.w;
  *(float4*)(y + (size_t)row * Dc + c) = yv;
}

// ===========================================================================
extern "C" void kernel_launch(void* const* d_in, const int* in_sizes, int n_in,
                              void* d_out, int out_size, void* d_ws,
                              size_t ws_size, hipStream_t stream) {
  const float* Q = (const float*)d_in[0];
  const float* K = (const float*)d_in[1];
  const float* V = (const float*)d_in[2];
  const float* Wq = (const float*)d_in[3];
  const float* Wk = (const float*)d_in[4];
  const float* Wv = (const float*)d_in[5];
  const float* Wo = (const float*)d_in[6];
  const float* gamma = (const float*)d_in[7];
  const float* beta = (const float*)d_in[8];

  char* w = (char*)d_ws;
  const size_t XP = (size_t)Bc * Sc * Dc * sizeof(bf16);  // 16 MB
  const size_t WP = (size_t)Dc * Dc * sizeof(bf16);       //  2 MB
  bf16* Qb = (bf16*)(w);
  bf16* Kb = (bf16*)(w + XP);
  bf16* Vb = (bf16*)(w + 2 * XP);
  bf16* Wqb = (bf16*)(w + 3 * XP);
  bf16* Wkb = (bf16*)(w + 3 * XP + WP);
  bf16* Wvb = (bf16*)(w + 3 * XP + 2 * WP);
  bf16* Wob = (bf16*)(w + 3 * XP + 3 * WP);
  bf16* qb = (bf16*)(w + 3 * XP + 4 * WP);
  bf16* kb = (bf16*)(w + 4 * XP + 4 * WP);
  bf16* vtb = (bf16*)(w + 5 * XP + 4 * WP);
  bf16* ab = (bf16*)(w + 6 * XP + 4 * WP);
  float* of = (float*)(w);  // reuses Qb/Kb region after proj_gemm is done

  const int XN = Bc * Sc * Dc / 8 / 256;  // 4096 blocks
  const int WN = Dc * Dc / 8 / 256;       //  512 blocks
  cvt_kernel<<<XN, 256, 0, stream>>>(Q, Qb);
  cvt_kernel<<<XN, 256, 0, stream>>>(K, Kb);
  cvt_kernel<<<XN, 256, 0, stream>>>(V, Vb);
  cvt_kernel<<<WN, 256, 0, stream>>>(Wq, Wqb);
  cvt_kernel<<<WN, 256, 0, stream>>>(Wk, Wkb);
  cvt_kernel<<<WN, 256, 0, stream>>>(Wv, Wvb);
  cvt_kernel<<<WN, 256, 0, stream>>>(Wo, Wob);

  proj_gemm<<<dim3(64, 8, 3), 256, 0, stream>>>(Qb, Kb, Vb, Wqb, Wkb, Wvb, qb,
                                                kb, vtb);
  attn_kernel<<<dim3(Bc * Hc * (Sc / 64)), 128, 0, stream>>>(qb, kb, vtb, ab);
  out_gemm<<<dim3(64, 8), 256, 0, stream>>>(ab, Wob, Q, of);
  ln_kernel<<<dim3(Bc * Sc), 256, 0, stream>>>(of, gamma, beta, (float*)d_out);
}
